// BertSelfAttention_48103633715360
// MI455X (gfx1250) — matrix operand order repaired and checked by reading, not compiled, not run on hardware
//
#include <hip/hip_runtime.h>

// Problem constants (match reference)
#define BB 4
#define SS 2048
#define DD 768
#define HH 12
#define DHD 64

typedef __bf16 bf16_t;
typedef __attribute__((ext_vector_type(16))) __bf16 v16bf;
typedef __attribute__((ext_vector_type(8)))  __bf16 bf16v8;
typedef __attribute__((ext_vector_type(8)))  float  v8f;

// ---- operand packing helpers -------------------------------------------------
// v16bf element i lives in VGPR i/2 (lo/hi 16b). Per CDNA5 ISA 7.12.2:
//   A (16x32 bf16): lane L holds row M=L&15, K = {k0..k0+7} U {k0+16..k0+23}, k0=(L>=16)*8
//   B (32x16 bf16): lane L holds col N=L&15, K = {k0..k0+7} U {k0+16..k0+23}, k0=(L>=16)*8
//   C/D (16x16 f32): VGPR r <-> row r+(L>=16)*8, lane <-> col L&15
static __device__ __forceinline__ v16bf pack16(bf16v8 lo, bf16v8 hi) {
    v16bf r;
#pragma unroll
    for (int i = 0; i < 8; ++i) { r[i] = lo[i]; r[i + 8] = hi[i]; }
    return r;
}
static __device__ __forceinline__ v16bf ldA_bf16(const bf16_t* row, int k0) {
    return pack16(*(const bf16v8*)(row + k0), *(const bf16v8*)(row + k0 + 16));
}
static __device__ __forceinline__ v16bf ldB_bf16(const bf16_t* col, int h16) {
    return pack16(*(const bf16v8*)(col + h16 / 2), *(const bf16v8*)(col + h16 / 2 + 16));
}

// ============================================================================
// Kernel 0: bulk f32 -> bf16 conversion (8 elems/thread, b128 in / b128 out).
// ============================================================================
__global__ __launch_bounds__(256) void cvt_f32_bf16_kernel(
    const float* __restrict__ src, bf16_t* __restrict__ dst, int n8)
{
    const int i = blockIdx.x * 256 + threadIdx.x;
    if (i >= n8) return;
    const float4 a = reinterpret_cast<const float4*>(src)[2 * i];
    const float4 b = reinterpret_cast<const float4*>(src)[2 * i + 1];
    bf16v8 r;
    r[0] = (__bf16)a.x; r[1] = (__bf16)a.y; r[2] = (__bf16)a.z; r[3] = (__bf16)a.w;
    r[4] = (__bf16)b.x; r[5] = (__bf16)b.y; r[6] = (__bf16)b.z; r[7] = (__bf16)b.w;
    reinterpret_cast<bf16v8*>(dst)[i] = r;
}

// ============================================================================
// Kernel 1: fused QKV projection, pure-bf16 GEMM, software-pipelined.
// Each wave computes a 16x64 output strip; iteration k+1's operands are
// loaded before iteration k's WMMAs so loads stay in flight during matrix ops.
// Q pre-scaled by 1/sqrt(DH); V stored transposed [B,H,DH,S].
// grid (8192/64, 12, 3), block 128 (4 waves).
// ============================================================================
__global__ __launch_bounds__(128) void qkv_gemm_kernel(
    const bf16_t* __restrict__ Xb,
    const bf16_t* __restrict__ Wqb, const float* __restrict__ bq,
    const bf16_t* __restrict__ Wkb, const float* __restrict__ bk,
    const bf16_t* __restrict__ Wvb, const float* __restrict__ bv,
    bf16_t* __restrict__ qo, bf16_t* __restrict__ ko, bf16_t* __restrict__ vto)
{
    const int lane = threadIdx.x & 31;
    const int wave = threadIdx.x >> 5;
    const int idx  = lane & 15;
    const int half = lane >> 4;
    const int k0   = half * 8;
    const int h16  = half * 16;

    const int z  = blockIdx.z;                    // 0=Q 1=K 2=V (uniform)
    const int hy = blockIdx.y;                    // head = 64-wide N strip
    const int mb = (blockIdx.x * 4 + wave) * 16;  // row base in [0, 8192)

    const bf16_t* W    = (z == 0) ? Wqb : (z == 1) ? Wkb : Wvb;
    const float*  bias = (z == 0) ? bq  : (z == 1) ? bk  : bv;

    const bf16_t* arow = Xb + (size_t)(mb + idx) * DD;
    const bf16_t* br0  = W + (size_t)(hy * 64 + idx) * DD;   // B col = W row

    v8f acc[4];
#pragma unroll
    for (int t = 0; t < 4; ++t) acc[t] = (v8f){0,0,0,0,0,0,0,0};

    // prologue: load iteration 0 operands
    v16bf a_cur = ldA_bf16(arow, k0);
    v16bf b_cur[4];
#pragma unroll
    for (int t = 0; t < 4; ++t) b_cur[t] = ldB_bf16(br0 + (size_t)(t * 16) * DD, h16);

#pragma unroll 2
    for (int kk = 0; kk < DD; kk += 32) {
        const int kn = (kk + 32 < DD) ? kk + 32 : 0;   // wraps on last iter (harmless)
        // issue next iteration's loads first ...
        v16bf a_nxt = ldA_bf16(arow + kn, k0);
        v16bf b_nxt[4];
#pragma unroll
        for (int t = 0; t < 4; ++t)
            b_nxt[t] = ldB_bf16(br0 + (size_t)(t * 16) * DD + kn, h16);
        // ... then consume current operands while those loads are in flight
#pragma unroll
        for (int t = 0; t < 4; ++t)
            acc[t] = __builtin_amdgcn_wmma_f32_16x16x32_bf16(
                false, a_cur, false, b_cur[t], (short)0, acc[t], false, false);
        a_cur = a_nxt;
#pragma unroll
        for (int t = 0; t < 4; ++t) b_cur[t] = b_nxt[t];
    }

    const float qscale = (z == 0) ? 0.125f : 1.0f;   // 1/sqrt(64)
#pragma unroll
    for (int t = 0; t < 4; ++t) {
        const int   dh   = t * 16 + idx;             // col within head
        const float bval = bias[hy * 64 + dh];
#pragma unroll
        for (int r = 0; r < 8; ++r) {
            const int   m = mb + r + half * 8;
            const int   b = m >> 11;                 // m / 2048
            const int   s = m & (SS - 1);
            const float v = (acc[t][r] + bval) * qscale;
            if (z == 2) {
                vto[(((size_t)b * HH + hy) * DHD + dh) * SS + s] = (__bf16)v;
            } else {
                bf16_t* dst = (z == 0) ? qo : ko;
                dst[(((size_t)b * HH + hy) * SS + s) * DHD + dh] = (__bf16)v;
            }
        }
    }
}

// ============================================================================
// Kernel 2: flash attention, 64 keys/iter.  All 8 K B-operands are loaded
// before the 8 QK^T WMMAs (one long load clause, waits amortized); likewise
// V B-operands are batched ahead of the P@V WMMAs and can issue during the
// softmax VALU work.  P converts C-layout -> A-layout through a 2KB per-wave
// LDS tile (same-wave DS ops are in-order).  grid (2048/64, 12, 4), block 128.
// ============================================================================
__global__ __launch_bounds__(128) void attn_kernel(
    const bf16_t* __restrict__ q, const bf16_t* __restrict__ kmat,
    const bf16_t* __restrict__ vt, const float* __restrict__ mask,
    float* __restrict__ out)
{
    __shared__ __bf16 ldsP[4][16 * 64];   // per-wave 16x64 P tile

    const int lane = threadIdx.x & 31;
    const int wave = threadIdx.x >> 5;
    const int idx  = lane & 15;
    const int half = lane >> 4;
    const int k0   = half * 8;
    const int h16  = half * 16;

    const int h     = blockIdx.y;
    const int b     = blockIdx.z;
    const int qbase = (blockIdx.x * 4 + wave) * 16;

    const bf16_t* qhead = q    + (((size_t)b * HH + h) * SS) * DHD;
    const bf16_t* khead = kmat + (((size_t)b * HH + h) * SS) * DHD;
    const bf16_t* vhead = vt   + (((size_t)b * HH + h) * DHD) * SS;
    const float*  mrow  = mask + (size_t)b * SS;

    // Q A-operands, held in registers for the whole key sweep
    const bf16_t* qrow = qhead + (size_t)(qbase + idx) * DHD;
    const v16bf aq0 = ldA_bf16(qrow, k0);        // dh 0..31
    const v16bf aq1 = ldA_bf16(qrow + 32, k0);   // dh 32..63

    float mrun[8], lrun[8];
    v8f o[4];
#pragma unroll
    for (int r = 0; r < 8; ++r) { mrun[r] = -3.0e38f; lrun[r] = 0.f; }
#pragma unroll
    for (int t = 0; t < 4; ++t) o[t] = (v8f){0,0,0,0,0,0,0,0};

    __bf16* Pl = ldsP[wave];

    for (int kb = 0; kb < SS; kb += 64) {
        // prefetch the next key/value block (global_prefetch_b8)
        const int kn = (kb + 64 < SS) ? kb + 64 : kb;
        __builtin_prefetch(khead + (size_t)(kn + idx) * DHD, 0, 3);
        __builtin_prefetch(vhead + (size_t)idx * SS + kn, 0, 3);

        // ---- batch-load all 8 K B-operands, then run the 8 QK^T WMMAs ----
        v16bf kb0[4], kb1[4];
#pragma unroll
        for (int g = 0; g < 4; ++g) {
            const bf16_t* kr = khead + (size_t)(kb + g * 16 + idx) * DHD;
            kb0[g] = ldB_bf16(kr, h16);        // dh 0..31
            kb1[g] = ldB_bf16(kr + 32, h16);   // dh 32..63
        }
        v8f c[4];
#pragma unroll
        for (int g = 0; g < 4; ++g) {
            v8f cc = (v8f){0,0,0,0,0,0,0,0};
            cc = __builtin_amdgcn_wmma_f32_16x16x32_bf16(
                false, aq0, false, kb0[g], (short)0, cc, false, false);
            cc = __builtin_amdgcn_wmma_f32_16x16x32_bf16(
                false, aq1, false, kb1[g], (short)0, cc, false, false);
            c[g] = cc;
        }

        // ---- batch-issue V B-operand loads (overlap with softmax below) ----
        v16bf vb0[4], vb1[4];
#pragma unroll
        for (int t = 0; t < 4; ++t) {
            const bf16_t* vrow = vhead + (size_t)(t * 16 + idx) * SS;
            vb0[t] = ldB_bf16(vrow + kb, h16);        // keys kb..+31
            vb1[t] = ldB_bf16(vrow + kb + 32, h16);   // keys +32..+63
        }

        float am[4];
#pragma unroll
        for (int g = 0; g < 4; ++g) am[g] = mrow[kb + g * 16 + idx];

        // ---- online softmax per row r (row = r + half*8) ----
#pragma unroll
        for (int r = 0; r < 8; ++r) {
            float s0 = c[0][r] + am[0];
            float s1 = c[1][r] + am[1];
            float s2 = c[2][r] + am[2];
            float s3 = c[3][r] + am[3];
            float mx = fmaxf(fmaxf(s0, s1), fmaxf(s2, s3));
#pragma unroll
            for (int off = 1; off < 16; off <<= 1)
                mx = fmaxf(mx, __shfl_xor(mx, off, 16));
            const float mnew = fmaxf(mrun[r], mx);
            const float scl  = __expf(mrun[r] - mnew);
            const float p0 = __expf(s0 - mnew);
            const float p1 = __expf(s1 - mnew);
            const float p2 = __expf(s2 - mnew);
            const float p3 = __expf(s3 - mnew);
            float ps = (p0 + p1) + (p2 + p3);
#pragma unroll
            for (int off = 1; off < 16; off <<= 1)
                ps += __shfl_xor(ps, off, 16);
            lrun[r] = lrun[r] * scl + ps;
            mrun[r] = mnew;
#pragma unroll
            for (int t = 0; t < 4; ++t) o[t][r] *= scl;   // rescale running O

            const int row = r + half * 8;
            Pl[row * 64 + idx]      = (__bf16)p0;          // C-layout -> LDS
            Pl[row * 64 + 16 + idx] = (__bf16)p1;
            Pl[row * 64 + 32 + idx] = (__bf16)p2;
            Pl[row * 64 + 48 + idx] = (__bf16)p3;
        }

        // ---- P out of LDS in A-layout: two 16x32 operands ----
        const v16bf ap0 = pack16(*(const bf16v8*)(Pl + idx * 64 + k0),
                                 *(const bf16v8*)(Pl + idx * 64 + k0 + 16));
        const v16bf ap1 = pack16(*(const bf16v8*)(Pl + idx * 64 + 32 + k0),
                                 *(const bf16v8*)(Pl + idx * 64 + 32 + k0 + 16));

        // ---- O += P @ V ----
#pragma unroll
        for (int t = 0; t < 4; ++t) {
            o[t] = __builtin_amdgcn_wmma_f32_16x16x32_bf16(
                false, ap0, false, vb0[t], (short)0, o[t], false, false);
            o[t] = __builtin_amdgcn_wmma_f32_16x16x32_bf16(
                false, ap1, false, vb1[t], (short)0, o[t], false, false);
        }
    }

    // ---- epilogue: normalize and store ctx[b][s][h*64+dh] ----
#pragma unroll
    for (int r = 0; r < 8; ++r) {
        const float inv = 1.0f / lrun[r];
        const int   s   = qbase + r + half * 8;
#pragma unroll
        for (int t = 0; t < 4; ++t) {
            const int d = h * DHD + t * 16 + idx;
            out[((size_t)b * SS + s) * DD + d] = o[t][r] * inv;
        }
    }
}

// ============================================================================
extern "C" void kernel_launch(void* const* d_in, const int* in_sizes, int n_in,
                              void* d_out, int out_size, void* d_ws, size_t ws_size,
                              hipStream_t stream) {
    const float* X    = (const float*)d_in[0];
    const float* mask = (const float*)d_in[1];
    const float* Wq   = (const float*)d_in[2];
    const float* bq   = (const float*)d_in[3];
    const float* Wk   = (const float*)d_in[4];
    const float* bk   = (const float*)d_in[5];
    const float* Wv   = (const float*)d_in[6];
    const float* bv   = (const float*)d_in[7];
    float* out = (float*)d_out;

    // workspace layout (bf16 elements):
    //   Xb [B*S, D], Wqb/Wkb/Wvb [D, D], Q [B,H,S,DH], K [B,H,S,DH], V^T [B,H,DH,S]
    const size_t nX = (size_t)BB * SS * DD;        // 6,291,456
    const size_t nW = (size_t)DD * DD;             //   589,824
    const size_t nP = (size_t)BB * HH * SS * DHD;  // 6,291,456
    bf16_t* xb  = (bf16_t*)d_ws;
    bf16_t* wqb = xb  + nX;
    bf16_t* wkb = wqb + nW;
    bf16_t* wvb = wkb + nW;
    bf16_t* q   = wvb + nW;
    bf16_t* k   = q   + nP;
    bf16_t* vt  = k   + nP;

    // 0) f32 -> bf16 conversions (vectorized, 8 elems/thread)
    cvt_f32_bf16_kernel<<<(int)(nX / 8 + 255) / 256, 256, 0, stream>>>(X,  xb,  (int)(nX / 8));
    cvt_f32_bf16_kernel<<<(int)(nW / 8 + 255) / 256, 256, 0, stream>>>(Wq, wqb, (int)(nW / 8));
    cvt_f32_bf16_kernel<<<(int)(nW / 8 + 255) / 256, 256, 0, stream>>>(Wk, wkb, (int)(nW / 8));
    cvt_f32_bf16_kernel<<<(int)(nW / 8 + 255) / 256, 256, 0, stream>>>(Wv, wvb, (int)(nW / 8));

    // 1) QKV projection GEMM (one 16x64 strip per wave, double-buffered)
    dim3 g1((BB * SS) / 64, HH, 3);   // (128, 12, 3)
    qkv_gemm_kernel<<<g1, 128, 0, stream>>>(xb, wqb, bq, wkb, bk, wvb, bv, q, k, vt);

    // 2) flash attention
    dim3 g2(SS / 64, HH, BB);         // (32, 12, 4)
    attn_kernel<<<g2, 128, 0, stream>>>(q, k, vt, mask, out);
}
